// GraspPoseNet_80401787781434
// MI455X (gfx1250) — compile-verified, matrix operand order checked by reading, not run on hardware
//
#include <hip/hip_runtime.h>

#define BB 4
#define NN 16384
#define PP 512
#define KK 100
#define KT 256   // K-chunk staged in LDS per block iteration

typedef __bf16 bf16_t;
typedef __bf16 v16bf __attribute__((ext_vector_type(16)));
typedef __bf16 v8bf  __attribute__((ext_vector_type(8)));
typedef float  v8f   __attribute__((ext_vector_type(8)));
typedef unsigned int u32x4 __attribute__((ext_vector_type(4)));
typedef int          i32x4 __attribute__((ext_vector_type(4)));
typedef int          i32x8 __attribute__((ext_vector_type(8)));

#ifndef HAVE_TDM
#if defined(__has_builtin)
#if __has_builtin(__builtin_amdgcn_tensor_load_to_lds)
#define HAVE_TDM 1
#endif
#endif
#endif
#ifndef HAVE_TDM
#define HAVE_TDM 0
#endif

// ============================================================================
// WMMA GEMM:  Y[n, m] = act( sum_k X[n,k] * A[m,k] + bias[m] )
//   A : (Mpad x Kpad) bf16 row-major, Mpad multiple of 64, zero-padded
//   X : (Nrows x ldX) bf16 row-major, cols >= Kpad, zero-padded
//   Y : (Nrows x ldY) bf16 or f32
// Block = 4 waves = 64(M) x 64(N) tile. The 64 x kc weight tile is staged in
// LDS once per K-chunk — via the Tensor Data Mover (tensor_load_to_lds +
// s_wait_tensorcnt) when available — and shared by all 4 waves. Each wave
// holds 4 accumulators (64M x 16N), so one B fragment feeds 4 WMMAs.
// A fragment (16-bit A 16x32): lane = row (lane&15); elem j<8 -> K =
// (lane>>4)*8 + j, j>=8 -> K = (lane>>4)*8 + 8 + j. B mirrors A with lane =
// column. D: VGPR r -> M = r + 8*(lane>>4), N = lane&15.
// ACT: 0 = linear, 1 = relu, 2 = sigmoid
// ============================================================================
template<int ACT, typename OutT>
__global__ __launch_bounds__(128) void gemm_bf16_kernel(
    const bf16_t* __restrict__ A, const bf16_t* __restrict__ X,
    const float* __restrict__ bias, OutT* __restrict__ Y,
    int ldY, int Mreal, int storeCols, int Kpad, int ldX, int Ntiles)
{
  __shared__ __align__(16) bf16_t shA[64 * KT];
  const int wave = threadIdx.x >> 5;
  const int lane = threadIdx.x & 31;
  int nt = blockIdx.y * 4 + wave;
  if (nt >= Ntiles) nt = Ntiles - 1;        // uniform per wave; duplicate work is benign
  const int mBase = blockIdx.x * 64;
  const int hi = lane >> 4;
  const int lm = lane & 15;
  const bf16_t* bp0 = X + (size_t)(nt * 16 + lm) * ldX + hi * 8;
  v8f acc[4] = {};

  for (int k0 = 0; k0 < Kpad; k0 += KT) {
    const int kc = (Kpad - k0) < KT ? (Kpad - k0) : KT;
    __syncthreads();                        // previous chunk fully consumed
#if HAVE_TDM
    if (wave == 0) {
      // Tensor DMA descriptor (D#): 2D tile, 64 rows x kc elems of 2-byte data.
      const unsigned long long ga =
          (unsigned long long)(const void*)(A + (size_t)mBase * Kpad + k0);
      u32x4 g0 = {};
      i32x8 g1 = {};
      i32x4 gz = {};
      g0[0] = 1u;                                             // count=1, user mode
      g0[1] = 0u;                                             // lds_addr: shA @ 0
      g0[2] = (unsigned)(ga & 0xffffffffu);                   // global_addr[31:0]
      g0[3] = (unsigned)((ga >> 32) & 0x01ffffffu)            // global_addr[56:32]
              | 0x80000000u;                                  // type=2 ("image")
      g1[0] = 1 << 16;                                        // data_size=1 (2B)
      g1[1] = (int)(((unsigned)Kpad & 0xffffu) << 16);        // tensor_dim0 lo16
      g1[2] = (int)((((unsigned)Kpad >> 16) & 0xffffu)        // tensor_dim0 hi16
              | (64u << 16));                                 // tensor_dim1 lo16 = 64
      g1[3] = (int)(((unsigned)kc & 0xffffu) << 16);          // tile_dim0
      g1[4] = 64;                                             // tile_dim1
      g1[5] = Kpad;                                           // tensor_dim0_stride lo32
      g1[6] = 0;
      g1[7] = 0;
#if __clang_major__ >= 23
      i32x8 gz8 = {};
      __builtin_amdgcn_tensor_load_to_lds(g0, g1, gz, gz, gz8, 0);
#else
      __builtin_amdgcn_tensor_load_to_lds(g0, g1, gz, gz, 0);
#endif
      __builtin_amdgcn_s_wait_tensorcnt(0);
    }
#else
    for (int idx = threadIdx.x; idx < 64 * (kc >> 3); idx += 128) {
      int r = idx / (kc >> 3), c8 = idx % (kc >> 3);
      *reinterpret_cast<v8bf*>(&shA[r * kc + c8 * 8]) =
          *reinterpret_cast<const v8bf*>(A + (size_t)(mBase + r) * Kpad + k0 + c8 * 8);
    }
#endif
    __syncthreads();                        // A tile visible to all waves

    const bf16_t* bp = bp0 + k0;
    for (int kk = 0; kk < kc; kk += 32) {
      v16bf bfrag;
      *reinterpret_cast<v8bf*>(&bfrag)       = *reinterpret_cast<const v8bf*>(bp);
      *(reinterpret_cast<v8bf*>(&bfrag) + 1) = *reinterpret_cast<const v8bf*>(bp + 16);
      __builtin_prefetch(bp + 128, 0, 1);   // global_prefetch_b8 ahead on X
#pragma unroll
      for (int mtl = 0; mtl < 4; ++mtl) {
        v16bf afrag;
        const bf16_t* apl = &shA[(size_t)(mtl * 16 + lm) * kc + kk + hi * 8];
        *reinterpret_cast<v8bf*>(&afrag)       = *reinterpret_cast<const v8bf*>(apl);
        *(reinterpret_cast<v8bf*>(&afrag) + 1) = *reinterpret_cast<const v8bf*>(apl + 16);
        acc[mtl] = __builtin_amdgcn_wmma_f32_16x16x32_bf16(false, afrag, false, bfrag,
                                                           (short)0, acc[mtl], false, false);
      }
      bp += 32;
    }
  }

  const int orow = nt * 16 + lm;
#pragma unroll
  for (int mtl = 0; mtl < 4; ++mtl) {
#pragma unroll
    for (int r = 0; r < 8; ++r) {
      int o = mBase + mtl * 16 + hi * 8 + r;        // M index of acc element r
      if (o >= storeCols) continue;
      float v = acc[mtl][r];
      if (o < Mreal) v += bias[o]; else v = 0.0f;   // zero the channel pads
      if (ACT == 1) v = v > 0.0f ? v : 0.0f;
      if (ACT == 2) v = 1.0f / (1.0f + __expf(-v));
      Y[(size_t)orow * ldY + o] = (OutT)v;
    }
  }
}

// fp32 weight (Cout x Kdim) -> bf16 (Mpad x Kpad) zero-padded
__global__ void convert_w_kernel(const float* __restrict__ W, bf16_t* __restrict__ Wb,
                                 int Cout, int Kdim, int Kpad, size_t total)
{
  size_t i = (size_t)blockIdx.x * blockDim.x + threadIdx.x;
  if (i >= total) return;
  int r = (int)(i / Kpad), c = (int)(i % Kpad);
  float v = (r < Cout && c < Kdim) ? W[(size_t)r * Kdim + c] : 0.0f;
  Wb[i] = (bf16_t)v;
}

// Farthest-point sampling: one workgroup per batch, iterative argmax reduction.
__global__ __launch_bounds__(1024) void fps_kernel(const float* __restrict__ xyz,
                                                   int n, int npoint, int* __restrict__ out)
{
  const int b = blockIdx.x;
  const int t = threadIdx.x;
  const float* p = xyz + (size_t)b * n * 3;
  __shared__ float sd[1024];
  __shared__ int   si[1024];
  float dist[16];
#pragma unroll
  for (int j = 0; j < 16; ++j) dist[j] = 1e10f;
  int sel = 0;
  for (int it = 0; it < npoint; ++it) {
    if (t == 0) out[b * npoint + it] = sel;
    float cx = p[sel*3+0], cy = p[sel*3+1], cz = p[sel*3+2];
    float best = -1.0f; int bi = 0x7fffffff;
    for (int j = 0; j < 16; ++j) {
      int i = t + j * 1024;
      if (i < n) {
        float dx = p[i*3+0]-cx, dy = p[i*3+1]-cy, dz = p[i*3+2]-cz;
        float d = fminf(dist[j], dx*dx+dy*dy+dz*dz);
        dist[j] = d;
        if (d > best) { best = d; bi = i; }   // strict > keeps lowest index on ties
      }
    }
    sd[t] = best; si[t] = bi;
    __syncthreads();
    for (int s = 512; s > 0; s >>= 1) {
      if (t < s) {
        if (sd[t+s] > sd[t] || (sd[t+s] == sd[t] && si[t+s] < si[t])) {
          sd[t] = sd[t+s]; si[t] = si[t+s];
        }
      }
      __syncthreads();
    }
    sel = si[0];
    __syncthreads();
  }
}

__global__ void gather_pts_kernel(const float* __restrict__ xyz, const int* __restrict__ idx,
                                  int n, int S, float* __restrict__ out, int total)
{
  int gid = blockIdx.x * blockDim.x + threadIdx.x;
  if (gid >= total) return;
  int b = gid / S;
  int i = idx[gid];
  out[gid*3+0] = xyz[((size_t)b*n + i)*3 + 0];
  out[gid*3+1] = xyz[((size_t)b*n + i)*3 + 1];
  out[gid*3+2] = xyz[((size_t)b*n + i)*3 + 2];
}

// Ball query (first ns indices in radius, pad with first hit / 0) + grouping:
// writes rows [(b*S+s)*ns+j] = [rel_xyz(3) | feats(Cf) | zero-pad..ldX]
__global__ void ball_group_kernel(const float* __restrict__ xyz, int n,
                                  const float* __restrict__ nxyz, int S,
                                  const bf16_t* __restrict__ feats, int Cf, int ldF,
                                  float r2, int ns, bf16_t* __restrict__ Xo, int ldX, int total)
{
  int gid = blockIdx.x * blockDim.x + threadIdx.x;
  if (gid >= total) return;
  int b = gid / S;
  const float* ctr = nxyz + (size_t)gid * 3;
  float cx = ctr[0], cy = ctr[1], cz = ctr[2];
  const float* pb = xyz + (size_t)b * n * 3;
  int found[32];
  int cnt = 0;
  for (int i = 0; i < n && cnt < ns; ++i) {
    float dx = pb[i*3+0]-cx, dy = pb[i*3+1]-cy, dz = pb[i*3+2]-cz;
    if (dx*dx + dy*dy + dz*dz < r2) found[cnt++] = i;
  }
  int first = (cnt > 0) ? found[0] : 0;
  for (int j = cnt; j < ns; ++j) found[j] = first;
  for (int j = 0; j < ns; ++j) {
    int i = found[j];
    bf16_t* o = Xo + ((size_t)gid * ns + j) * ldX;
    o[0] = (bf16_t)(pb[i*3+0] - cx);
    o[1] = (bf16_t)(pb[i*3+1] - cy);
    o[2] = (bf16_t)(pb[i*3+2] - cz);
    if (Cf > 0) {
      const bf16_t* f = feats + ((size_t)b * n + i) * ldF;
      for (int c = 0; c < Cf; ++c) o[3 + c] = f[c];
    }
    for (int c = 3 + Cf; c < ldX; ++c) o[c] = (bf16_t)0.0f;
  }
}

// max over ns consecutive rows, write to concat destination at column offset
__global__ void maxpool_concat_kernel(const bf16_t* __restrict__ Yin, int ns, int C, int ldIn,
                                      bf16_t* __restrict__ Fo, int ldOut, int coff, int Gtot)
{
  int gid = blockIdx.x * blockDim.x + threadIdx.x;
  if (gid >= Gtot * C) return;
  int g = gid / C, c = gid % C;
  float m = -3.0e38f;
  for (int j = 0; j < ns; ++j)
    m = fmaxf(m, (float)Yin[((size_t)g * ns + j) * ldIn + c]);
  Fo[(size_t)g * ldOut + coff + c] = (bf16_t)m;
}

// 3-NN inverse-distance interpolation + optional channel concat of uf.
__global__ void fp_interp_kernel(const float* __restrict__ ux, int nU,
                                 const float* __restrict__ kx, int nK,
                                 const bf16_t* __restrict__ kf, int ldK, int Ck,
                                 const bf16_t* __restrict__ uf, int ldU, int Cu,
                                 bf16_t* __restrict__ Xo, int ldX, int total)
{
  int gid = blockIdx.x * blockDim.x + threadIdx.x;
  if (gid >= total) return;
  int b = gid / nU;
  float px = ux[gid*3+0], py = ux[gid*3+1], pz = ux[gid*3+2];
  const float* kb = kx + (size_t)b * nK * 3;
  float d0 = 3e38f, d1 = 3e38f, d2 = 3e38f;
  int i0 = 0, i1 = 0, i2 = 0;
  for (int i = 0; i < nK; ++i) {
    float dx = kb[i*3+0]-px, dy = kb[i*3+1]-py, dz = kb[i*3+2]-pz;
    float d = dx*dx + dy*dy + dz*dz;
    if      (d < d0) { d2=d1; i2=i1; d1=d0; i1=i0; d0=d; i0=i; }
    else if (d < d1) { d2=d1; i2=i1; d1=d; i1=i; }
    else if (d < d2) { d2=d; i2=i; }
  }
  float w0 = 1.0f / (fmaxf(d0, 0.0f) + 1e-8f);
  float w1 = 1.0f / (fmaxf(d1, 0.0f) + 1e-8f);
  float w2 = 1.0f / (fmaxf(d2, 0.0f) + 1e-8f);
  float s = w0 + w1 + w2; w0 /= s; w1 /= s; w2 /= s;
  const bf16_t* f0 = kf + ((size_t)b * nK + i0) * ldK;
  const bf16_t* f1 = kf + ((size_t)b * nK + i1) * ldK;
  const bf16_t* f2 = kf + ((size_t)b * nK + i2) * ldK;
  bf16_t* o = Xo + (size_t)gid * ldX;
  for (int c = 0; c < Ck; ++c)
    o[c] = (bf16_t)(w0 * (float)f0[c] + w1 * (float)f1[c] + w2 * (float)f2[c]);
  if (uf) {
    const bf16_t* u = uf + (size_t)gid * ldU;
    for (int c = 0; c < Cu; ++c) o[Ck + c] = u[c];
  }
  for (int c = Ck + Cu; c < ldX; ++c) o[c] = (bf16_t)0.0f;
}

// Head gather: Xh[(b*P+p), c*K + k] = feat[b*N + lp[b,p,k], c]
// (contraction index c*K+k matches Conv2d weight (o, c, k) flattening)
__global__ void gather_head_kernel(const bf16_t* __restrict__ feat, int ldF,
                                   const int* __restrict__ lp, bf16_t* __restrict__ Xh)
{
  int gid = blockIdx.x * blockDim.x + threadIdx.x;
  if (gid >= BB * PP * KK) return;
  int bp = gid / KK, k = gid % KK;
  int b = bp / PP;
  int idx = lp[gid];
  const bf16_t* f = feat + ((size_t)b * NN + idx) * ldF;
  bf16_t* o = Xh + (size_t)bp * (128 * KK) + k;
  for (int c = 0; c < 128; ++c) o[(size_t)c * KK] = f[c];
}

// ============================================================================
// Host orchestration
// ============================================================================
extern "C" void kernel_launch(void* const* d_in, const int* in_sizes, int n_in,
                              void* d_out, int out_size, void* d_ws, size_t ws_size,
                              hipStream_t stream)
{
  (void)in_sizes; (void)n_in; (void)out_size; (void)ws_size;
  const float* pc = (const float*)d_in[0];
  const int*   lp = (const int*)d_in[1];

  // params pytree leaves (nested dicts flatten in sorted-key order: ap, fp, pose, sa;
  // each conv leaf pair is (w, b)).
  enum {
    AP_C1W = 2, AP_C1B, AP_S1W, AP_S1B, AP_S2W, AP_S2B, AP_WMW, AP_WMB,
    FP_BASE = 10,     // fp[i] layer j: w = FP_BASE + (i*2+j)*2
    POSE_BASE = 26,   // a1w,a1b,a2w,a2b,c1w,c1b,o1w,o1b,o2w,o2b,s1w,s1b,s2w,s2b,wmw,wmb
    SA_BASE = 42      // sa[L][br] layer ly: w = SA_BASE + ((L*2+br)*3+ly)*2
  };

  char* wsbase = (char*)d_ws;
  size_t wsoff = 0;
  auto balloc = [&](size_t bytes) -> void* {
    size_t o = (wsoff + 255) & ~(size_t)255;
    wsoff = o + bytes;
    return (void*)(wsbase + o);
  };
  auto rup = [](int x, int m) { return ((x + m - 1) / m) * m; };

  struct WB { bf16_t* w; const float* bias; int Mreal, Mpad, Kpad; };
  auto convw = [&](int wi, int bi, int Cout, int Kdim) -> WB {
    WB r; r.Mreal = Cout; r.Mpad = rup(Cout, 64); r.Kpad = rup(Kdim, 32);
    size_t tot = (size_t)r.Mpad * r.Kpad;
    r.w = (bf16_t*)balloc(tot * sizeof(bf16_t));
    r.bias = (const float*)d_in[bi];
    convert_w_kernel<<<dim3((unsigned)((tot + 255) / 256)), 256, 0, stream>>>(
        (const float*)d_in[wi], r.w, Cout, Kdim, r.Kpad, tot);
    return r;
  };
  auto gemm_bf = [&](const WB& w, const bf16_t* X, int ldX, int Nrows, bf16_t* Y, int ldY) {
    int Ntiles = Nrows / 16;
    dim3 grid((unsigned)((ldY + 63) / 64), (unsigned)((Ntiles + 3) / 4));
    gemm_bf16_kernel<1, bf16_t><<<grid, 128, 0, stream>>>(
        w.w, X, w.bias, Y, ldY, w.Mreal, ldY, w.Kpad, ldX, Ntiles);
  };
  auto gemm_f32 = [&](const WB& w, const bf16_t* X, int ldX, int Nrows,
                      float* Y, int ldY, int storeCols, int act) {
    int Ntiles = Nrows / 16;
    dim3 grid((unsigned)((storeCols + 63) / 64), (unsigned)((Ntiles + 3) / 4));
    if (act == 2)
      gemm_bf16_kernel<2, float><<<grid, 128, 0, stream>>>(
          w.w, X, w.bias, Y, ldY, w.Mreal, storeCols, w.Kpad, ldX, Ntiles);
    else
      gemm_bf16_kernel<0, float><<<grid, 128, 0, stream>>>(
          w.w, X, w.bias, Y, ldY, w.Mreal, storeCols, w.Kpad, ldX, Ntiles);
  };

  // ---------------- weight conversion ----------------
  static const int saC[4][2][4] = {
    { {  3,  16,  16,  32}, {  3,  32,  32,  64} },
    { { 99,  64,  64, 128}, { 99,  64,  96, 128} },
    { {259, 128, 196, 256}, {259, 128, 196, 256} },
    { {515, 256, 256, 512}, {515, 256, 384, 512} } };
  WB saw[4][2][3];
  for (int L = 0; L < 4; ++L)
    for (int br = 0; br < 2; ++br)
      for (int ly = 0; ly < 3; ++ly) {
        int base = SA_BASE + ((L * 2 + br) * 3 + ly) * 2;
        saw[L][br][ly] = convw(base, base + 1, saC[L][br][ly + 1], saC[L][br][ly]);
      }
  static const int fpC[4][3] = { {256,128,128}, {608,256,256}, {768,512,512}, {1536,512,512} };
  WB fpw[4][2];
  for (int i = 0; i < 4; ++i)
    for (int j = 0; j < 2; ++j) {
      int base = FP_BASE + (i * 2 + j) * 2;
      fpw[i][j] = convw(base, base + 1, fpC[i][j + 1], fpC[i][j]);
    }
  WB ap_wm = convw(AP_WMW, AP_WMB, 128, 128 * KK);
  WB ap_c1 = convw(AP_C1W, AP_C1B, 512, 128);
  WB ap_s1 = convw(AP_S1W, AP_S1B, 256, 512);
  WB ap_s2 = convw(AP_S2W, AP_S2B, 1, 256);
  WB po_a1 = convw(POSE_BASE + 0,  POSE_BASE + 1,  256, 512);
  WB po_a2 = convw(POSE_BASE + 2,  POSE_BASE + 3,  1,   256);
  WB po_c1 = convw(POSE_BASE + 4,  POSE_BASE + 5,  512, 128);
  WB po_o1 = convw(POSE_BASE + 6,  POSE_BASE + 7,  256, 512);
  WB po_o2 = convw(POSE_BASE + 8,  POSE_BASE + 9,  3,   256);
  WB po_wm = convw(POSE_BASE + 14, POSE_BASE + 15, 128, 128 * KK);

  // ---------------- scratch buffers ----------------
  static const int npts[5] = { NN, 1024, 256, 64, 16 };
  static const int ldf[5]  = { 0, 96, 256, 512, 1024 };
  static const float rad[4][2] = { {0.05f,0.1f}, {0.1f,0.2f}, {0.2f,0.4f}, {0.4f,0.8f} };
  static const int nsm[2] = { 16, 32 };

  int* fidx = (int*)balloc(sizeof(int) * BB * 1024);
  float* nx[5]; nx[0] = const_cast<float*>(pc);
  for (int l = 1; l < 5; ++l) nx[l] = (float*)balloc(sizeof(float) * (size_t)BB * npts[l] * 3);
  bf16_t* lf[5]; lf[0] = nullptr;
  for (int l = 1; l < 5; ++l) lf[l] = (bf16_t*)balloc(sizeof(bf16_t) * (size_t)BB * npts[l] * ldf[l]);
  bf16_t* Xg  = (bf16_t*)balloc(sizeof(bf16_t) * (size_t)4194304);  // max grouped X
  bf16_t* Ya  = (bf16_t*)balloc(sizeof(bf16_t) * (size_t)8388608);  // GEMM ping
  bf16_t* Yb  = (bf16_t*)balloc(sizeof(bf16_t) * (size_t)8388608);  // GEMM pong
  bf16_t* fpX = (bf16_t*)balloc(sizeof(bf16_t) * (size_t)BB * NN * 256);
  bf16_t* nf[4];
  nf[0] = (bf16_t*)balloc(sizeof(bf16_t) * (size_t)BB * NN * 128);
  nf[1] = (bf16_t*)balloc(sizeof(bf16_t) * (size_t)BB * 1024 * 256);
  nf[2] = (bf16_t*)balloc(sizeof(bf16_t) * (size_t)BB * 256 * 512);
  nf[3] = (bf16_t*)balloc(sizeof(bf16_t) * (size_t)BB * 64 * 512);
  bf16_t* Xh  = (bf16_t*)balloc(sizeof(bf16_t) * (size_t)BB * PP * 128 * KK);
  bf16_t* f0h = (bf16_t*)balloc(sizeof(bf16_t) * (size_t)BB * PP * 128);
  bf16_t* f1h = (bf16_t*)balloc(sizeof(bf16_t) * (size_t)BB * PP * 512);
  bf16_t* f2h = (bf16_t*)balloc(sizeof(bf16_t) * (size_t)BB * PP * 256);

  // ---------------- SA backbone ----------------
  for (int L = 0; L < 4; ++L) {
    int nPrev = npts[L], S = npts[L + 1];
    fps_kernel<<<BB, 1024, 0, stream>>>(nx[L], nPrev, S, fidx);
    gather_pts_kernel<<<(BB * S + 255) / 256, 256, 0, stream>>>(nx[L], fidx, nPrev, S,
                                                                nx[L + 1], BB * S);
    int coff = 0;
    for (int br = 0; br < 2; ++br) {
      int ns = nsm[br];
      int Cin0 = saC[L][br][0];
      int Cf = Cin0 - 3;
      int ldX = rup(Cin0, 32);
      float r = rad[L][br];
      ball_group_kernel<<<(BB * S + 127) / 128, 128, 0, stream>>>(
          nx[L], nPrev, nx[L + 1], S, lf[L], Cf, ldf[L], r * r, ns, Xg, ldX, BB * S);
      int rows = BB * S * ns;
      const bf16_t* curX = Xg; int cLd = ldX;
      bf16_t* dst = Ya;
      for (int ly = 0; ly < 3; ++ly) {
        const WB& w = saw[L][br][ly];
        gemm_bf(w, curX, cLd, rows, dst, w.Mpad);
        curX = dst; cLd = w.Mpad;
        dst = (dst == Ya) ? Yb : Ya;
      }
      int Cout = saC[L][br][3];
      maxpool_concat_kernel<<<(BB * S * Cout + 255) / 256, 256, 0, stream>>>(
          curX, ns, Cout, cLd, lf[L + 1], ldf[L + 1], coff, BB * S);
      coff += Cout;
    }
  }

  // ---------------- FP (feature propagation) ----------------
  bf16_t* curF[5]; int curLd[5], curC[5];
  curF[0] = nullptr; curLd[0] = 0; curC[0] = 0;
  for (int l = 1; l < 5; ++l) { curF[l] = lf[l]; curLd[l] = ldf[l]; curC[l] = ldf[l]; }
  for (int i = 3; i >= 0; --i) {
    int nU = npts[i], nK = npts[i + 1];
    int Ck = curC[i + 1], Cu = (i > 0) ? curC[i] : 0;
    int ldX = rup(Ck + Cu, 32);
    int total = BB * nU;
    fp_interp_kernel<<<(total + 127) / 128, 128, 0, stream>>>(
        nx[i], nU, nx[i + 1], nK, curF[i + 1], curLd[i + 1], Ck,
        (i > 0) ? curF[i] : (const bf16_t*)nullptr, (i > 0) ? curLd[i] : 0, Cu,
        fpX, ldX, total);
    gemm_bf(fpw[i][0], fpX, ldX, total, Ya, fpw[i][0].Mpad);
    gemm_bf(fpw[i][1], Ya, fpw[i][0].Mpad, total, nf[i], fpw[i][1].Mpad);
    curF[i] = nf[i]; curC[i] = fpC[i][2]; curLd[i] = fpw[i][1].Mpad;
  }

  // ---------------- heads ----------------
  gather_head_kernel<<<(BB * PP * KK + 255) / 256, 256, 0, stream>>>(nf[0], curLd[0], lp, Xh);
  float* outp = (float*)d_out;
  const int rowsH = BB * PP;                    // 2048 rows = (b, p)
  // ap head: score = sigmoid(s2(s1(c1(wm(gf)))))
  gemm_bf (ap_wm, Xh,  128 * KK, rowsH, f0h, ap_wm.Mpad);
  gemm_bf (ap_c1, f0h, ap_wm.Mpad, rowsH, f1h, ap_c1.Mpad);
  gemm_bf (ap_s1, f1h, ap_c1.Mpad, rowsH, f2h, ap_s1.Mpad);
  gemm_f32(ap_s2, f2h, ap_s1.Mpad, rowsH, outp, 1, 1, 2);             // (B,P) sigmoid
  // pose head: offset = o2(o1(trunk)), angle = a2(a1(trunk))  (score discarded)
  gemm_bf (po_wm, Xh,  128 * KK, rowsH, f0h, po_wm.Mpad);
  gemm_bf (po_c1, f0h, po_wm.Mpad, rowsH, f1h, po_c1.Mpad);
  gemm_bf (po_o1, f1h, po_c1.Mpad, rowsH, f2h, po_o1.Mpad);
  gemm_f32(po_o2, f2h, po_o1.Mpad, rowsH, outp + rowsH, 3, 3, 0);     // (B,P,3) linear
  gemm_bf (po_a1, f1h, po_c1.Mpad, rowsH, f2h, po_a1.Mpad);
  gemm_f32(po_a2, f2h, po_a1.Mpad, rowsH, outp + rowsH * 4, 1, 1, 0); // (B,P) linear
}
